// MultiModalCrossAttentionConcatFusion_28132035788930
// MI455X (gfx1250) — matrix operand order repaired and checked by reading, not compiled, not run on hardware
//
#include <hip/hip_runtime.h>
#include <hip/hip_bf16.h>

#if __has_include(<hip/amd_detail/amd_gfx1250_TDM.h>)
#define ATHENA_TDM_6ARG 1
#endif

// ---------------------------------------------------------------------------
// MI455X (gfx1250) fused implementation.
//  - softmax over singleton axis == identity  ->  attn == vh  (q/k heads dead)
//  - dominant FLOPs: lin1/lin2 (77 of 85 GFLOP) -> v_wmma_f32_16x16x32_f16
//  - FF weights DMA'd per-workgroup into LDS with the Tensor Data Mover
//    (double buffered, TENSORcnt synchronized) instead of per-wave vmem.
// ---------------------------------------------------------------------------

typedef __attribute__((ext_vector_type(16))) _Float16     v16h;
typedef __attribute__((ext_vector_type(8)))  float        v8f;
typedef __attribute__((ext_vector_type(4)))  unsigned int v4u;
typedef __attribute__((ext_vector_type(4)))  unsigned int u32x4;
typedef __attribute__((ext_vector_type(8)))  int          i32x8;
typedef __attribute__((ext_vector_type(4)))  int          i32x4;

#define B_ROWS 32768
#define DMODEL 144
#define DPAD   160      // K padded to multiple of 32 for 16x16x32 WMMA
#define FFDIM  2048

#define XF_S 145        // f32 X stride (floats)
#define XS_S 168        // f16 X stride (halves): 84 DW % 64 = 20 -> conflict-free b128
#define HS_S 136        // f16 h-chunk stride  : 68 DW % 64 = 4  -> conflict-free b128
#define VS_S 168        // f16 v stride (lives in chunk buffer 0)

#define CHUNK_N   128               // FF features per chunk
#define NCHUNK    16                // 2048 / 128
#define W1_LD     168               // halves per W1 row in blob
#define W2_LD     136               // halves per W2 row in blob
#define W1_HALVES (128 * W1_LD)     // 21504
#define W2_HALVES (144 * W2_LD)     // 19584
#define HB_HALVES (W1_HALVES + W2_HALVES)   // 41088 halves = 82176 B = 10272 x 8B
#define HB_UNITS  10272             // blob size in 8-byte TDM units

// LDS layout (bytes) inside encoder_kernel dynamic segment
#define LDS_XF   0
#define LDS_XS   (LDS_XF + 128 * XF_S * 4)        // 74240
#define LDS_HS   (LDS_XS + 128 * XS_S * 2)        // 117248
#define LDS_RED  (LDS_HS + 128 * HS_S * 2)        // 152064
#define LDS_BUF0 (LDS_RED + 512 * 4)              // 154112
#define LDS_BUF1 (LDS_BUF0 + HB_HALVES * 2)       // 236288
#define LDS_END  (LDS_BUF1 + HB_HALVES * 2)       // 318464  (<= 320KB)

// ---- WMMA fragment loaders (CDNA5 16-bit A / B layouts, wave32) -----------
// A (16x32, MxK): lane ln is row M; lanes 0-15 hold K[0..7]+K[16..23],
// lanes 16-31 hold K[8..15]+K[24..31] (packed pairs per VGPR).
__device__ __forceinline__ v16h load_a16(const _Float16* base, int stride,
                                         int row0, int k0, int ln, int lhi) {
  union { v16h v; v4u u[2]; } r;
  const _Float16* p = base + (row0 + ln) * stride + k0 + 8 * lhi;
  r.u[0] = *(const v4u*)(p);
  r.u[1] = *(const v4u*)(p + 16);
  return r.v;
}
// B (32x16, KxN): lane ln is column N; lanes 0-15 hold K[0..7]+K[16..23], lanes 16-31
// hold K[8..15]+K[24..31]. Weights stored (out,in) row-major == B^T.
__device__ __forceinline__ v16h load_b16(const _Float16* W, int ldk,
                                         int n0, int k0, int ln, int lhi) {
  union { v16h v; v4u u[2]; } r;
  const _Float16* p = W + (size_t)(n0 + ln) * ldk + k0 + 8 * lhi;
  r.u[0] = *(const v4u*)(p);
  r.u[1] = *(const v4u*)(p + 16);
  return r.v;
}
#define WMMA_F16(a, b, c) \
  __builtin_amdgcn_wmma_f32_16x16x32_f16(false, (a), false, (b), (short)0, (c), false, false)

// ---- Tensor Data Mover: DMA one contiguous weight blob (82,176 B) to LDS ---
// D# per CDNA5 ISA: group0 = {count, lds_addr, gaddr, type},
// group1 = dims/strides in data_size units (8 B). 1-row 2D tile.
__device__ __forceinline__ void tdm_load_blob(unsigned lds_off, const void* gptr) {
  unsigned long long ga = (unsigned long long)gptr;
  u32x4 g0 = { 1u,                                   // count=1 (valid), user mode
               lds_off,                               // lds_addr (bytes)
               (unsigned)ga,                          // global_addr[31:0]
               (unsigned)((ga >> 32) & 0x01FFFFFFu) | 0x80000000u }; // [56:32]|type=2
  i32x8 g1 = { 0x00030000,                            // data_size=3 (8B), mask=0
               (int)((HB_UNITS & 0xFFFF) << 16),      // tensor_dim0[15:0]
               (int)((HB_UNITS >> 16) | (1 << 16)),   // tensor_dim0[31:16]|tensor_dim1 lo
               (int)(HB_UNITS << 16),                 // tile_dim0
               1,                                     // tile_dim1 = 1
               HB_UNITS,                              // tensor_dim0_stride[31:0]
               0, 0 };
  i32x4 z4 = {0, 0, 0, 0};
#ifdef ATHENA_TDM_6ARG
  i32x8 z8 = {0, 0, 0, 0, 0, 0, 0, 0};
  __builtin_amdgcn_tensor_load_to_lds(g0, g1, z4, z4, z8, 0);
#else
  __builtin_amdgcn_tensor_load_to_lds(g0, g1, z4, z4, 0);
#endif
}

// ---------------------------------------------------------------------------
// Weight conversion fp32 -> f16 with K padding (src stride == cols).
// ---------------------------------------------------------------------------
__global__ __launch_bounds__(256) void cvt_pad_kernel(const float* __restrict__ src,
                                                      _Float16* __restrict__ dst,
                                                      int rows, int cols, int colsPad) {
  int idx = blockIdx.x * 256 + threadIdx.x;
  if (idx >= rows * colsPad) return;
  int r = idx / colsPad, c = idx - r * colsPad;
  dst[idx] = (c < cols) ? (_Float16)src[(size_t)r * cols + c] : (_Float16)0.f;
}

// Pack FF weights into per-(layer,chunk) TDM blobs:
//  [ W1 rows fc*128..+128, 168 halves each (K 0..143 data, 144..167 zero) |
//    W2 rows 0..143, cols fc*128..+128, 136 halves each (128 data + 8 zero) ]
__global__ __launch_bounds__(256) void build_blob_kernel(const float* __restrict__ lin1w,
                                                         const float* __restrict__ lin2w,
                                                         _Float16* __restrict__ dst) {
  long long idx = (long long)blockIdx.x * 256 + threadIdx.x;
  if (idx >= (long long)2 * NCHUNK * HB_HALVES) return;
  int blob = (int)(idx / HB_HALVES);
  int w    = (int)(idx - (long long)blob * HB_HALVES);
  int l = blob >> 4, fc = blob & 15;
  float v = 0.f;
  if (w < W1_HALVES) {
    int r = w / W1_LD, c = w - r * W1_LD;
    if (c < DMODEL)
      v = lin1w[((size_t)l * FFDIM + fc * CHUNK_N + r) * DMODEL + c];
  } else {
    int w2 = w - W1_HALVES;
    int n = w2 / W2_LD, c = w2 - n * W2_LD;
    if (c < CHUNK_N)
      v = lin2w[((size_t)l * DMODEL + n) * FFDIM + fc * CHUNK_N + c];
  }
  dst[idx] = (_Float16)v;
}

// ---------------------------------------------------------------------------
// Phase A: embeddings + pair cross (softmax identity) -> X (B x 144, f32)
// ---------------------------------------------------------------------------
struct BuildArgs {
  const float* x[4];
  const float* Wq[4];  const float* bq[4];
  const float* Wkv[4]; const float* bkv[4];
  const float* aiw; const float* aib; const float* aow; const float* aob;
  float* X;
};

__global__ __launch_bounds__(256) void build_x_kernel(BuildArgs A) {
  extern __shared__ char smem0[];
  float* s_xout = (float*)smem0;            // 256 x 148
  float* s_kv   = s_xout + 256 * 148;       // 256 x 37
  float* s_tile = s_kv   + 256 * 37;        // 256 x 33
  float* s_wq   = s_tile + 256 * 33;        // 9 x 32
  float* s_wkv  = s_wq   + 9 * 32;          // 9 x 32

  const int tid  = threadIdx.x;
  const int row0 = blockIdx.x * 256;

  #pragma unroll
  for (int m = 0; m < 4; m++) {
    const int Fm = (m == 0) ? 128 : (m == 1) ? 256 : (m == 2) ? 64 : 512;
    const float* xm = A.x[m] + (size_t)row0 * Fm;
    float qa[9], ka[9];
    #pragma unroll
    for (int e = 0; e < 9; e++) { qa[e] = 0.f; ka[e] = 0.f; }

    for (int c0 = 0; c0 < Fm; c0 += 32) {
      for (int idx = tid; idx < 256 * 32; idx += 256) {
        int r = idx >> 5, c = idx & 31;
        s_tile[r * 33 + c] = xm[(size_t)r * Fm + c0 + c];   // coalesced
      }
      for (int idx = tid; idx < 9 * 32; idx += 256) {
        int e = idx >> 5, c = idx & 31;
        s_wq[idx]  = A.Wq[m][e * Fm + c0 + c];
        s_wkv[idx] = A.Wkv[m][e * Fm + c0 + c];
      }
      __syncthreads();
      const float* tr = s_tile + tid * 33;
      #pragma unroll
      for (int c = 0; c < 32; c++) {
        float xv = tr[c];
        #pragma unroll
        for (int e = 0; e < 9; e++) {
          qa[e] = fmaf(s_wq[e * 32 + c], xv, qa[e]);
          ka[e] = fmaf(s_wkv[e * 32 + c], xv, ka[e]);
        }
      }
      __syncthreads();
    }
    #pragma unroll
    for (int e = 0; e < 9; e++) {
      s_xout[tid * 148 + m * 36 + e] = qa[e] + A.bq[m][e];
      s_kv[tid * 37 + m * 9 + e]     = ka[e] + A.bkv[m][e];
    }
  }

  // Pair cross: attn == vh because softmax over a length-1 axis is 1.0.
  const float* kvr = s_kv + tid * 37;
  for (int i = 0; i < 4; i++) {
    for (int jj = 0; jj < 3; jj++) {
      int j = (jj < i) ? jj : jj + 1;
      int p = i * 3 + jj;
      float vh[9];
      #pragma unroll
      for (int e = 0; e < 9; e++) {
        float acc = A.aib[p * 27 + 18 + e];
        #pragma unroll
        for (int k = 0; k < 9; k++)
          acc = fmaf(A.aiw[(p * 27 + 18 + e) * 9 + k], kvr[j * 9 + k], acc);
        vh[e] = acc;
      }
      #pragma unroll
      for (int ff = 0; ff < 9; ff++) {
        float acc = A.aob[p * 9 + ff];
        #pragma unroll
        for (int e = 0; e < 9; e++)
          acc = fmaf(A.aow[(p * 9 + ff) * 9 + e], vh[e], acc);
        s_xout[tid * 148 + i * 36 + 9 + jj * 9 + ff] = acc;
      }
    }
  }
  __syncthreads();
  for (int idx = tid; idx < 256 * 144; idx += 256) {    // fully contiguous store
    int r = idx / 144, c = idx - r * 144;
    A.X[(size_t)row0 * 144 + idx] = s_xout[r * 148 + c];
  }
}

// ---------------------------------------------------------------------------
// Phase B: both encoder layers fused; WG owns 128 rows; FF weights via TDM.
// ---------------------------------------------------------------------------
struct EncArgs {
  const float* X;
  const _Float16* Wv; const _Float16* Wo; const _Float16* blobs;
  const float* enc_in_b; const float* enc_out_b; const float* lin1_b; const float* lin2_b;
  const float* ln1w; const float* ln1b; const float* ln2w; const float* ln2b;
  float* out;
};

__device__ __forceinline__ void layernorm_block(float* Xf, _Float16* Xs,
                                                float* redS, float* redQ,
                                                const float* g, const float* b, int tid) {
  const int row = tid >> 1, part = tid & 1, c0 = part * 72;
  float* xr = Xf + row * XF_S + c0;
  float s = 0.f, q = 0.f;
  for (int c = 0; c < 72; c++) { float v = xr[c]; s += v; q = fmaf(v, v, q); }
  redS[tid] = s; redQ[tid] = q;
  __syncthreads();
  float S = redS[row * 2] + redS[row * 2 + 1];
  float Q = redQ[row * 2] + redQ[row * 2 + 1];
  float mean = S * (1.f / 144.f);
  float var  = Q * (1.f / 144.f) - mean * mean;
  float inv  = rsqrtf(var + 1e-5f);
  _Float16* xh = Xs + row * XS_S + c0;
  for (int c = 0; c < 72; c++) {
    float v = (xr[c] - mean) * inv * g[c0 + c] + b[c0 + c];
    xr[c] = v; xh[c] = (_Float16)v;
  }
  __syncthreads();
}

__global__ __launch_bounds__(256) void encoder_kernel(EncArgs A) {
  extern __shared__ char smem1[];
  float*    Xf   = (float*)(smem1 + LDS_XF);
  _Float16* Xs   = (_Float16*)(smem1 + LDS_XS);
  _Float16* Hs   = (_Float16*)(smem1 + LDS_HS);
  float*    redS = (float*)(smem1 + LDS_RED);
  float*    redQ = redS + 256;
  _Float16* buf0 = (_Float16*)(smem1 + LDS_BUF0);
  _Float16* buf1 = (_Float16*)(smem1 + LDS_BUF1);
  // byte offsets within the wave's LDS allocation (generic addr low 32 bits)
  const unsigned buf0_off = (unsigned)(unsigned long long)(void*)buf0;
  const unsigned buf1_off = (unsigned)(unsigned long long)(void*)buf1;

  const int tid = threadIdx.x, wave = tid >> 5, lane = tid & 31;
  const int ln = lane & 15, lhi = lane >> 4;
  const int m0 = wave * 16;
  const int row0 = blockIdx.x * 128;

  const v8f vzero = {0.f, 0.f, 0.f, 0.f, 0.f, 0.f, 0.f, 0.f};

  for (int idx = tid; idx < 128 * 144; idx += 256) {
    int r = idx / 144, c = idx - r * 144;
    float v = A.X[(size_t)row0 * 144 + idx];
    Xf[r * XF_S + c] = v;
    Xs[r * XS_S + c] = (_Float16)v;
  }
  for (int idx = tid; idx < 128 * 16; idx += 256) {   // zero K padding 144..159
    int r = idx >> 4, c = 144 + (idx & 15);
    Xs[r * XS_S + c] = (_Float16)0.f;
  }
  __syncthreads();

  for (int l = 0; l < 2; l++) {
    const _Float16* Wv = A.Wv + l * 144 * DPAD;
    const _Float16* Wo = A.Wo + l * 144 * DPAD;
    const _Float16* blobL = A.blobs + (size_t)l * NCHUNK * HB_HALVES;
    const float* bv = A.enc_in_b + l * 432 + 288;
    const float* bo = A.enc_out_b + l * 144;
    const float* b1 = A.lin1_b + l * FFDIM;
    const float* b2 = A.lin2_b + l * 144;
    _Float16* Vs = buf0;            // v matrix parks in chunk buffer 0

    // ---- v = X @ Wv^T + bv  -> Vs (f16) --------------------------------
    {
      v16h a5[5];
      #pragma unroll
      for (int kt = 0; kt < 5; kt++) a5[kt] = load_a16(Xs, XS_S, m0, kt * 32, ln, lhi);
      #pragma unroll
      for (int nt = 0; nt < 9; nt++) {
        v8f c = vzero;
        #pragma unroll
        for (int kt = 0; kt < 5; kt++)
          c = WMMA_F16(a5[kt], load_b16(Wv, DPAD, nt * 16, kt * 32, ln, lhi), c);
        int n = nt * 16 + ln;
        float bb = bv[n];
        #pragma unroll
        for (int r = 0; r < 8; r++)
          Vs[(m0 + r + 8 * lhi) * VS_S + n] = (_Float16)(c[r] + bb);
      }
      #pragma unroll
      for (int r = 0; r < 8; r++)                      // zero K pad of v
        Vs[(m0 + r + 8 * lhi) * VS_S + 144 + ln] = (_Float16)0.f;
    }
    __syncthreads();

    // ---- sa = v @ Wo^T + bo; Xf += sa ----------------------------------
    {
      v16h a5[5];
      #pragma unroll
      for (int kt = 0; kt < 5; kt++) a5[kt] = load_a16(Vs, VS_S, m0, kt * 32, ln, lhi);
      #pragma unroll
      for (int nt = 0; nt < 9; nt++) {
        v8f c = vzero;
        #pragma unroll
        for (int kt = 0; kt < 5; kt++)
          c = WMMA_F16(a5[kt], load_b16(Wo, DPAD, nt * 16, kt * 32, ln, lhi), c);
        int n = nt * 16 + ln;
        float bb = bo[n];
        #pragma unroll
        for (int r = 0; r < 8; r++)
          Xf[(m0 + r + 8 * lhi) * XF_S + n] += c[r] + bb;
      }
    }
    __syncthreads();
    // kick off DMA of FF chunk 0 while LayerNorm runs (Vs fully consumed)
    if (wave == 0) tdm_load_blob(buf0_off, blobL);
    layernorm_block(Xf, Xs, redS, redQ, A.ln1w + l * 144, A.ln1b + l * 144, tid);

    // ---- FF: acc = relu(X@W1^T + b1) @ W2^T, TDM double-buffered -------
    v8f acc[9];
    #pragma unroll
    for (int i = 0; i < 9; i++) acc[i] = vzero;

    v16h ax[5];                     // X fragments are invariant across FF
    #pragma unroll
    for (int kt = 0; kt < 5; kt++) ax[kt] = load_a16(Xs, XS_S, m0, kt * 32, ln, lhi);

    for (int fc = 0; fc < NCHUNK; fc++) {
      const _Float16* bufc = (fc & 1) ? buf1 : buf0;
      if (wave == 0) {
        if (fc + 1 < NCHUNK) {
          tdm_load_blob((fc & 1) ? buf0_off : buf1_off,
                        blobL + (size_t)(fc + 1) * HB_HALVES);
          __builtin_amdgcn_s_wait_tensorcnt(1);   // chunk fc complete
        } else {
          __builtin_amdgcn_s_wait_tensorcnt(0);   // last chunk: drain all
        }
      }
      __syncthreads();              // chunk fc resident for all waves

      // produce h chunk (128 x 128) -> Hs, B frags from LDS W1 part
      #pragma unroll
      for (int nt = 0; nt < 8; nt++) {
        v8f c = vzero;
        #pragma unroll
        for (int kt = 0; kt < 5; kt++)
          c = WMMA_F16(ax[kt], load_b16(bufc, W1_LD, nt * 16, kt * 32, ln, lhi), c);
        int n = nt * 16 + ln;
        float bb = b1[fc * CHUNK_N + n];
        #pragma unroll
        for (int r = 0; r < 8; r++) {
          float v = c[r] + bb;
          v = v > 0.f ? v : 0.f;                       // relu
          Hs[(m0 + r + 8 * lhi) * HS_S + n] = (_Float16)v;
        }
      }
      __syncthreads();

      // consume: acc += h @ W2_chunk^T (K = 128), B frags from LDS W2 part
      {
        const _Float16* W2p = bufc + W1_HALVES;
        v16h ah[4];
        #pragma unroll
        for (int kt = 0; kt < 4; kt++) ah[kt] = load_a16(Hs, HS_S, m0, kt * 32, ln, lhi);
        #pragma unroll
        for (int nt = 0; nt < 9; nt++) {
          #pragma unroll
          for (int kt = 0; kt < 4; kt++)
            acc[nt] = WMMA_F16(ah[kt], load_b16(W2p, W2_LD, nt * 16, kt * 32, ln, lhi), acc[nt]);
        }
      }
      __syncthreads();
    }
    #pragma unroll
    for (int nt = 0; nt < 9; nt++) {
      int n = nt * 16 + ln;
      float bb = b2[n];
      #pragma unroll
      for (int r = 0; r < 8; r++)
        Xf[(m0 + r + 8 * lhi) * XF_S + n] += acc[nt][r] + bb;
    }
    __syncthreads();
    layernorm_block(Xf, Xs, redS, redQ, A.ln2w + l * 144, A.ln2b + l * 144, tid);
  }

  for (int idx = tid; idx < 128 * 144; idx += 256) {
    int r = idx / 144, c = idx - r * 144;
    A.out[(size_t)row0 * 144 + idx] = Xf[r * XF_S + c];
  }
}

// ---------------------------------------------------------------------------
extern "C" void kernel_launch(void* const* d_in, const int* in_sizes, int n_in,
                              void* d_out, int out_size, void* d_ws, size_t ws_size,
                              hipStream_t stream) {
  (void)in_sizes; (void)out_size; (void)ws_size;
  static const int F[4] = {128, 256, 64, 512};

  const float* x[4]; const float* Wq[4]; const float* bq[4];
  const float* Wkv[4]; const float* bkv[4];
  for (int m = 0; m < 4; m++) x[m] = (const float*)d_in[m];

  int base;
  if (n_in >= 30) {             // tuples expanded to separate inputs
    for (int m = 0; m < 4; m++) {
      Wq[m]  = (const float*)d_in[4 + m];
      bq[m]  = (const float*)d_in[8 + m];
      Wkv[m] = (const float*)d_in[12 + m];
      bkv[m] = (const float*)d_in[16 + m];
    }
    base = 20;
  } else {                      // tuples flattened/concatenated per dict key
    const float* wqc = (const float*)d_in[4];
    const float* bqc = (const float*)d_in[5];
    const float* wkc = (const float*)d_in[6];
    const float* bkc = (const float*)d_in[7];
    int off = 0;
    for (int m = 0; m < 4; m++) {
      Wq[m] = wqc + off; Wkv[m] = wkc + off;
      bq[m] = bqc + m * 9; bkv[m] = bkc + m * 9;
      off += 9 * F[m];
    }
    base = 8;
  }
  const float* aiw = (const float*)d_in[base + 0];
  const float* aib = (const float*)d_in[base + 1];
  const float* aow = (const float*)d_in[base + 2];
  const float* aob = (const float*)d_in[base + 3];
  const float* enc_in_w  = (const float*)d_in[base + 4];
  const float* enc_in_b  = (const float*)d_in[base + 5];
  const float* enc_out_w = (const float*)d_in[base + 6];
  const float* enc_out_b = (const float*)d_in[base + 7];
  const float* ln1w = (const float*)d_in[base + 8];
  const float* ln1b = (const float*)d_in[base + 9];
  const float* lin1w = (const float*)d_in[base + 10];
  const float* lin1b = (const float*)d_in[base + 11];
  const float* lin2w = (const float*)d_in[base + 12];
  const float* lin2b = (const float*)d_in[base + 13];
  const float* ln2w = (const float*)d_in[base + 14];
  const float* ln2b = (const float*)d_in[base + 15];

  // workspace: X (f32) | Wv16 | Wo16 | FF weight blobs  (~21.7 MB)
  float* Xbuf = (float*)d_ws;
  char* p = (char*)d_ws + (size_t)B_ROWS * 144 * 4;
  _Float16* Wv16  = (_Float16*)p; p += (size_t)2 * 144 * DPAD * 2;
  _Float16* Wo16  = (_Float16*)p; p += (size_t)2 * 144 * DPAD * 2;
  _Float16* Blobs = (_Float16*)p;

  for (int l = 0; l < 2; l++) {
    int n = 144 * DPAD;
    cvt_pad_kernel<<<(n + 255) / 256, 256, 0, stream>>>(
        enc_in_w + (size_t)(l * 432 + 288) * 144, Wv16 + (size_t)l * 144 * DPAD, 144, 144, DPAD);
    cvt_pad_kernel<<<(n + 255) / 256, 256, 0, stream>>>(
        enc_out_w + (size_t)l * 144 * 144, Wo16 + (size_t)l * 144 * DPAD, 144, 144, DPAD);
  }
  {
    long long n = (long long)2 * NCHUNK * HB_HALVES;
    build_blob_kernel<<<(unsigned)((n + 255) / 256), 256, 0, stream>>>(lin1w, lin2w, Blobs);
  }

  BuildArgs ba;
  for (int m = 0; m < 4; m++) {
    ba.x[m] = x[m]; ba.Wq[m] = Wq[m]; ba.bq[m] = bq[m];
    ba.Wkv[m] = Wkv[m]; ba.bkv[m] = bkv[m];
  }
  ba.aiw = aiw; ba.aib = aib; ba.aow = aow; ba.aob = aob; ba.X = Xbuf;
  size_t smemA = (256 * 148 + 256 * 37 + 256 * 33 + 9 * 32 * 2) * sizeof(float);
  build_x_kernel<<<B_ROWS / 256, 256, smemA, stream>>>(ba);

  EncArgs ea;
  ea.X = Xbuf; ea.Wv = Wv16; ea.Wo = Wo16; ea.blobs = Blobs;
  ea.enc_in_b = enc_in_b; ea.enc_out_b = enc_out_b;
  ea.lin1_b = lin1b; ea.lin2_b = lin2b;
  ea.ln1w = ln1w; ea.ln1b = ln1b; ea.ln2w = ln2w; ea.ln2b = ln2b;
  ea.out = (float*)d_out;
  encoder_kernel<<<B_ROWS / 128, 256, LDS_END, stream>>>(ea);
}
